// VqVae_12867722019111
// MI455X (gfx1250) — matrix operand order repaired and checked by reading, not compiled, not run on hardware
//
#include <hip/hip_runtime.h>
#include <hip/hip_bf16.h>

// ---------- types ----------
typedef __attribute__((ext_vector_type(16))) __bf16        v16bf;
typedef __attribute__((ext_vector_type(8)))  float         v8f;
typedef __attribute__((ext_vector_type(4)))  unsigned int  u32x4;

#define BM 128
#define BN 128
#define BK 32
#define LDA 40   // padded LDS row stride (bf16 elements)
#define LDB 40

__device__ __forceinline__ unsigned short f32_to_bf16(float f) {
  unsigned int u = __float_as_uint(f);
  u += 0x7FFFu + ((u >> 16) & 1u);          // round-to-nearest-even
  return (unsigned short)(u >> 16);
}

// ---------- f32 -> bf16 convert (plain) ----------
__global__ void k_cvt_bf16(const float* __restrict__ src,
                           unsigned short* __restrict__ dst, int n) {
  int i = blockIdx.x * blockDim.x + threadIdx.x;
  if (i < n) dst[i] = f32_to_bf16(src[i]);
}

// ---------- f32 [K][N] -> bf16 [N][K] (weight transpose, once per launch) ----------
__global__ void k_cvt_bf16_t(const float* __restrict__ src,
                             unsigned short* __restrict__ dst, int K, int N) {
  int i = blockIdx.x * blockDim.x + threadIdx.x;
  if (i < K * N) {
    int k = i / N, n = i - k * N;
    dst[(size_t)n * K + k] = f32_to_bf16(src[i]);
  }
}

// ---------- tiled bf16 WMMA GEMM:  O = act(A[M,K] @ Bt^T + bias) ----------
// A  : [M][K] bf16 row-major
// Bt : [N][K] bf16 row-major (pre-transposed weights / codebook)
// Block tile 128x128, 8 waves, wave tile 64x32 (4x2 WMMAs/K-step), LDS double-buffered.
__global__ __launch_bounds__(256)
void k_gemm_bf16(const unsigned short* __restrict__ A,
                 const unsigned short* __restrict__ Bt,
                 const float* __restrict__ bias,
                 unsigned short* __restrict__ Obf,
                 float* __restrict__ Of,
                 int M, int N, int K, int relu)
{
  __shared__ __align__(16) unsigned short Asm[2][BM * LDA];
  __shared__ __align__(16) unsigned short Bsm[2][BN * LDB];

  const int tid  = threadIdx.x;
  const int lane = tid & 31;
  const int wave = tid >> 5;       // 8 waves
  const int wm   = wave >> 2;      // 0..1  (64 rows each)
  const int wn   = wave & 3;       // 0..3  (32 cols each)
  const int l16  = lane & 15;
  const int hsel = lane >> 4;      // half-wave select

  const int mblk = blockIdx.y * BM;
  const int nblk = blockIdx.x * BN;

  // staging: 128 rows x 32 k = 4096 bf16 per matrix; thread covers rows r0 and r0+64
  const int r0 = tid >> 2;
  const int q0 = (tid & 3) * 8;    // k-offset in elements

  const unsigned short* gA = A  + (size_t)mblk * K;
  const unsigned short* gB = Bt + (size_t)nblk * K;

  // ---- prologue: stage tile k0=0 into buffer 0 ----
  {
    u32x4 a0 = *(const u32x4*)(gA + (size_t)r0 * K + q0);
    u32x4 a1 = *(const u32x4*)(gA + (size_t)(r0 + 64) * K + q0);
    u32x4 b0 = *(const u32x4*)(gB + (size_t)r0 * K + q0);
    u32x4 b1 = *(const u32x4*)(gB + (size_t)(r0 + 64) * K + q0);
    *(u32x4*)(&Asm[0][r0 * LDA + q0])        = a0;
    *(u32x4*)(&Asm[0][(r0 + 64) * LDA + q0]) = a1;
    *(u32x4*)(&Bsm[0][r0 * LDB + q0])        = b0;
    *(u32x4*)(&Bsm[0][(r0 + 64) * LDB + q0]) = b1;
  }

  v8f acc[4][2] = {};
  const unsigned short* aCur = Asm[0];
  const unsigned short* bCur = Bsm[0];
  unsigned short* aNxt = Asm[1];
  unsigned short* bNxt = Bsm[1];

  for (int k0 = 0; k0 < K; k0 += BK) {
    __syncthreads();

    // prefetch next tile into registers while computing this one
    const bool pf = (k0 + BK) < K;
    u32x4 a0, a1, b0, b1;
    if (pf) {
      const int kk = k0 + BK;
      a0 = *(const u32x4*)(gA + (size_t)r0 * K + kk + q0);
      a1 = *(const u32x4*)(gA + (size_t)(r0 + 64) * K + kk + q0);
      b0 = *(const u32x4*)(gB + (size_t)r0 * K + kk + q0);
      b1 = *(const u32x4*)(gB + (size_t)(r0 + 64) * K + kk + q0);
    }

    // ---- fragments per ISA 16-bit layouts ----
    v16bf afr[4], bfr[2];
#pragma unroll
    for (int i = 0; i < 4; ++i) {
      // A 16x32: lane<16 -> K 0-7 / 16-23 ; lane>=16 -> K 8-15 / 24-31
      const unsigned short* p = &aCur[(wm * 64 + i * 16 + l16) * LDA + hsel * 8];
      union { u32x4 u[2]; v16bf v; } t;
      t.u[0] = *(const u32x4*)(p);
      t.u[1] = *(const u32x4*)(p + 16);
      afr[i] = t.v;
    }
#pragma unroll
    for (int j = 0; j < 2; ++j) {
      // B 32x16: lane<16 -> K 0-7 / 16-23 ; lane>=16 -> K 8-15 / 24-31
      const unsigned short* p = &bCur[(wn * 32 + j * 16 + l16) * LDB + hsel * 8];
      union { u32x4 u[2]; v16bf v; } t;
      t.u[0] = *(const u32x4*)(p);
      t.u[1] = *(const u32x4*)(p + 16);
      bfr[j] = t.v;
    }

#pragma unroll
    for (int i = 0; i < 4; ++i)
#pragma unroll
      for (int j = 0; j < 2; ++j)
        acc[i][j] = __builtin_amdgcn_wmma_f32_16x16x32_bf16(
            false, afr[i], false, bfr[j], (short)0, acc[i][j], false, false);

    if (pf) {
      *(u32x4*)(&aNxt[r0 * LDA + q0])        = a0;
      *(u32x4*)(&aNxt[(r0 + 64) * LDA + q0]) = a1;
      *(u32x4*)(&bNxt[r0 * LDB + q0])        = b0;
      *(u32x4*)(&bNxt[(r0 + 64) * LDB + q0]) = b1;
    }
    // swap ping-pong buffers
    const unsigned short* ta = aCur; aCur = aNxt; aNxt = (unsigned short*)ta;
    const unsigned short* tb = bCur; bCur = bNxt; bNxt = (unsigned short*)tb;
  }

  // ---- epilogue: bias + relu, write bf16 and/or f32 ----
#pragma unroll
  for (int j = 0; j < 2; ++j) {
    int n = nblk + wn * 32 + j * 16 + l16;
    float bv = bias ? bias[n] : 0.0f;
#pragma unroll
    for (int i = 0; i < 4; ++i) {
      int mbase = mblk + wm * 64 + i * 16 + hsel * 8;  // VGPR e -> row e (+8 upper half)
#pragma unroll
      for (int e = 0; e < 8; ++e) {
        float v = acc[i][j][e] + bv;
        if (relu) v = fmaxf(v, 0.0f);
        size_t off = (size_t)(mbase + e) * (size_t)N + (size_t)n;
        if (Of)  Of[off]  = v;
        if (Obf) Obf[off] = f32_to_bf16(v);
      }
    }
  }
}

// ---------- codebook norms + zero loss slot ----------
__global__ void k_cnorm(const float* __restrict__ cb, float* __restrict__ cnorm,
                        float* __restrict__ loss_slot, int D) {
  int row  = blockIdx.x * 8 + (threadIdx.x >> 5);
  int lane = threadIdx.x & 31;
  const float* r = cb + (size_t)row * D;
  float s = 0.f;
  for (int i = lane; i < D; i += 32) { float v = r[i]; s += v * v; }
#pragma unroll
  for (int o = 16; o > 0; o >>= 1) s += __shfl_xor(s, o, 32);
  if (lane == 0) cnorm[row] = s;
  if (blockIdx.x == 0 && threadIdx.x == 0) *loss_slot = 0.0f;
}

// ---------- per-row argmin + gather + loss ----------
__global__ void k_argmin(const float* __restrict__ scores,   // [B, Kc]  (z . c)
                         const float* __restrict__ cnorm,    // [Kc]
                         const float* __restrict__ zf,       // [B, D] f32
                         const float* __restrict__ cb,       // [Kc, D] f32
                         float* __restrict__ q_out,          // [B, D] f32 (d_out)
                         unsigned short* __restrict__ q_bf,  // [B, D] bf16 (next GEMM in)
                         float* __restrict__ idx_out,        // [B] (as float)
                         float* __restrict__ loss_slot,
                         int Bn, int Kc, int D)
{
  int row  = blockIdx.x * 8 + (threadIdx.x >> 5);
  int lane = threadIdx.x & 31;
  const float* s = scores + (size_t)row * Kc;
  float best = 3.402823466e38f; int bi = 0x7fffffff;
  for (int n = lane; n < Kc; n += 32) {
    float d = cnorm[n] - 2.0f * s[n];       // ||c||^2 - 2 z.c  (+||z||^2 const)
    if (d < best) { best = d; bi = n; }
  }
#pragma unroll
  for (int o = 16; o > 0; o >>= 1) {        // min, first-index tie-break
    float ob = __shfl_xor(best, o, 32);
    int   oi = __shfl_xor(bi,   o, 32);
    if (ob < best || (ob == best && oi < bi)) { best = ob; bi = oi; }
  }
  const float* c = cb + (size_t)bi * D;
  const float* z = zf + (size_t)row * D;
  float ss = 0.f;
  for (int i = lane; i < D; i += 32) {
    float q = c[i];
    float d = q - z[i];
    ss += d * d;
    q_out[(size_t)row * D + i] = q;                 // quantized_st (== quantized fwd)
    q_bf [(size_t)row * D + i] = f32_to_bf16(q);
  }
#pragma unroll
  for (int o = 16; o > 0; o >>= 1) ss += __shfl_xor(ss, o, 32);
  if (lane == 0) {
    idx_out[row] = (float)bi;
    atomicAdd(loss_slot, 1.25f * ss / ((float)Bn * (float)D)); // q_loss + 0.25*e_loss
  }
}

// ---------- host orchestration ----------
extern "C" void kernel_launch(void* const* d_in, const int* in_sizes, int n_in,
                              void* d_out, int out_size, void* d_ws, size_t ws_size,
                              hipStream_t stream) {
  (void)in_sizes; (void)n_in; (void)out_size; (void)ws_size;
  const int Bb = 16384, Din = 1024, H1 = 2048, H2 = 1024, Dc = 256, Kc = 1024;

  const float* x   = (const float*)d_in[0];
  const float* eW1 = (const float*)d_in[1];  const float* eb1 = (const float*)d_in[2];
  const float* eW2 = (const float*)d_in[3];  const float* eb2 = (const float*)d_in[4];
  const float* eW3 = (const float*)d_in[5];  const float* eb3 = (const float*)d_in[6];
  const float* cb  = (const float*)d_in[7];
  const float* dW1 = (const float*)d_in[8];  const float* db1 = (const float*)d_in[9];
  const float* dW2 = (const float*)d_in[10]; const float* db2 = (const float*)d_in[11];
  const float* dW3 = (const float*)d_in[12]; const float* db3 = (const float*)d_in[13];

  // workspace carve-up (256B aligned)
  char* ws = (char*)d_ws; size_t off = 0;
  auto alloc = [&](size_t bytes) {
    char* p = ws + off; off = (off + bytes + 255) & ~(size_t)255; return p;
  };
  // all weights stored transposed: [N][K] bf16
  unsigned short* we1 = (unsigned short*)alloc((size_t)Din*H1*2);  // [H1][Din]
  unsigned short* we2 = (unsigned short*)alloc((size_t)H1*H2*2);   // [H2][H1]
  unsigned short* we3 = (unsigned short*)alloc((size_t)H2*Dc*2);   // [Dc][H2]
  unsigned short* wcb = (unsigned short*)alloc((size_t)Kc*Dc*2);   // [Kc][Dc] (native)
  unsigned short* wd1 = (unsigned short*)alloc((size_t)Dc*H2*2);   // [H2][Dc]
  unsigned short* wd2 = (unsigned short*)alloc((size_t)H2*H1*2);   // [H1][H2]
  unsigned short* wd3 = (unsigned short*)alloc((size_t)H1*Din*2);  // [Din][H1]
  char*  act0 = alloc((size_t)Bb*H1*2);   // 64MB: xbf / H2 bf16 / scores f32 / dec-h1 bf16
  char*  act1 = alloc((size_t)Bb*H1*2);   // 64MB: H1 bf16 / z bf16 / q bf16 / dec-h2 bf16
  float* zf   = (float*)alloc((size_t)Bb*Dc*4);
  float* cn   = (float*)alloc((size_t)Kc*4);

  // d_out layout: reconstructed | quantized_st | loss | indices
  float* rec_out  = (float*)d_out;
  float* q_out    = rec_out + (size_t)Bb*Din;
  float* loss_out = q_out   + (size_t)Bb*Dc;
  float* idx_out  = loss_out + 1;

  auto cvt = [&](const float* s, unsigned short* d, int n) {
    k_cvt_bf16<<<(n + 255) / 256, 256, 0, stream>>>(s, d, n);
  };
  auto cvtT = [&](const float* s, unsigned short* d, int K, int N) {
    k_cvt_bf16_t<<<(K * N + 255) / 256, 256, 0, stream>>>(s, d, K, N);
  };
  cvtT(eW1, we1, Din, H1); cvtT(eW2, we2, H1, H2); cvtT(eW3, we3, H2, Dc);
  cvt (cb,  wcb, Kc * Dc);
  cvtT(dW1, wd1, Dc, H2);  cvtT(dW2, wd2, H2, H1); cvtT(dW3, wd3, H1, Din);
  cvt (x, (unsigned short*)act0, Bb * Din);

  auto gemm = [&](const void* A, const void* Bw, const float* bias,
                  void* Obf, float* Of, int M, int N, int K, int relu) {
    dim3 grid(N / BN, M / BM);
    k_gemm_bf16<<<grid, 256, 0, stream>>>((const unsigned short*)A,
        (const unsigned short*)Bw, bias, (unsigned short*)Obf, Of,
        M, N, K, relu);
  };

  // encoder
  gemm(act0, we1, eb1, act1, nullptr, Bb, H1, Din, 1);   // x -> h1 (relu)
  gemm(act1, we2, eb2, act0, nullptr, Bb, H2, H1,  1);   // h1 -> h2 (relu)
  gemm(act0, we3, eb3, act1, zf,      Bb, Dc, H2,  0);   // h2 -> z (bf16 + f32)

  // vector quantization
  k_cnorm<<<Kc / 8, 256, 0, stream>>>(cb, cn, loss_out, Dc);
  gemm(act1, wcb, nullptr, nullptr, (float*)act0, Bb, Kc, Dc, 0); // z . c^T
  k_argmin<<<Bb / 8, 256, 0, stream>>>((const float*)act0, cn, zf, cb,
      q_out, (unsigned short*)act1, idx_out, loss_out, Bb, Kc, Dc);

  // decoder
  gemm(act1, wd1, db1, act0, nullptr, Bb, H2, Dc, 1);    // q -> h (relu)
  gemm(act0, wd2, db2, act1, nullptr, Bb, H1, H2, 1);    // h -> h (relu)
  gemm(act1, wd3, db3, nullptr, rec_out, Bb, Din, H1, 0);// h -> reconstructed (f32)
}
